// Decoder_32418413150538
// MI455X (gfx1250) — matrix operand order repaired and checked by reading, not compiled, not run on hardware
//
#include <hip/hip_runtime.h>
#include <cstdint>
#include <cstddef>

// ---------------------------------------------------------------------------
// Types for CDNA5 WMMA (wave32): D(16x16,f32) = A(16x32,f16) x B(32x16,f16) + C
// ---------------------------------------------------------------------------
typedef _Float16 h16;
typedef __attribute__((ext_vector_type(8)))  _Float16 v8h;
typedef __attribute__((ext_vector_type(16))) _Float16 v16h;
typedef __attribute__((ext_vector_type(8)))  float    v8f;

__device__ __forceinline__ v16h cat16(v8h lo, v8h hi) {
    v16h r;
#pragma unroll
    for (int i = 0; i < 8; ++i) { r[i] = lo[i]; r[8 + i] = hi[i]; }
    return r;
}

__device__ __forceinline__ v8f wmma_f16(v16h a, v16h b, v8f c) {
    return __builtin_amdgcn_wmma_f32_16x16x32_f16(false, a, false, b, (short)0, c, false, false);
}

// ---------------------------------------------------------------------------
// Problem constants
// ---------------------------------------------------------------------------
#define NB    16
#define SEQ   1024
#define HID   512
#define HEADS 8
#define DH    64
#define PF    2048
#define ROWS  (NB * SEQ)          // 16384

// ---------------------------------------------------------------------------
// f32 -> f16 convert
// ---------------------------------------------------------------------------
__global__ void cvt_f32_f16(const float* __restrict__ in, h16* __restrict__ out, size_t n) {
    size_t i = (size_t)blockIdx.x * blockDim.x + threadIdx.x;
    size_t stride = (size_t)gridDim.x * blockDim.x;
    for (; i < n; i += stride) out[i] = (h16)in[i];
}

// ---------------------------------------------------------------------------
// WMMA GEMM: OUT[M,N] = act(A[M,K] @ W[N,K]^T + bias[N])
// Block: 256 threads (8 waves), tile 128x128, BK=32.
// Wave w: m-offset (w&1)*64, n-offset (w>>1)*32 -> 4x2 accumulators of 16x16.
// LDS pitch 40 halves (80B): conflict-free ds_load_b128 fragment reads.
// Requires M%128==0, N%128==0, K%32==0 (true for all uses here).
// ---------------------------------------------------------------------------
#define GBM 128
#define GBK 32
#define LDP 40

__global__ __launch_bounds__(256) void wmma_gemm(
    const h16* __restrict__ A, const h16* __restrict__ W,
    const float* __restrict__ bias,
    h16* __restrict__ out16, float* __restrict__ out32,
    int M, int N, int K, int relu)
{
    __shared__ __align__(16) h16 sA[GBM * LDP];
    __shared__ __align__(16) h16 sW[GBM * LDP];

    const int tid  = threadIdx.x;
    const int lane = tid & 31;
    const int wid  = tid >> 5;
    const int half = lane >> 4;
    const int l16  = lane & 15;

    const int gm0 = blockIdx.y * GBM;
    const int gn0 = blockIdx.x * GBM;
    const int wm  = (wid & 1) * 64;
    const int wn  = (wid >> 1) * 32;

    v8f acc[4][2] = {};

    // cooperative global->LDS mapping: 2 threads per 32-half row
    const int lrow  = tid >> 1;
    const int lpart = (tid & 1) * 16;
    const h16* gA = A + (size_t)(gm0 + lrow) * K + lpart;
    const h16* gW = W + (size_t)(gn0 + lrow) * K + lpart;
    h16* sAp = sA + lrow * LDP + lpart;
    h16* sWp = sW + lrow * LDP + lpart;

    for (int k0 = 0; k0 < K; k0 += GBK) {
        *(v8h*)(sAp)     = *(const v8h*)(gA + k0);
        *(v8h*)(sAp + 8) = *(const v8h*)(gA + k0 + 8);
        *(v8h*)(sWp)     = *(const v8h*)(gW + k0);
        *(v8h*)(sWp + 8) = *(const v8h*)(gW + k0 + 8);
        __syncthreads();

        // B-fragments: lane l16 = column n, k = half*8 + {0..7, 16..23}
        v16h bf[2];
#pragma unroll
        for (int ni = 0; ni < 2; ++ni) {
            const v8h* p = (const v8h*)(sW + (wn + ni * 16 + l16) * LDP + half * 8);
            bf[ni] = cat16(p[0], p[2]);
        }
        // A-fragments + 8 WMMAs
#pragma unroll
        for (int mi = 0; mi < 4; ++mi) {
            const h16* p = sA + (wm + mi * 16 + l16) * LDP + half * 8;
            v16h af = cat16(*(const v8h*)p, *(const v8h*)(p + 16));
            acc[mi][0] = wmma_f16(af, bf[0], acc[mi][0]);
            acc[mi][1] = wmma_f16(af, bf[1], acc[mi][1]);
        }
        __syncthreads();
    }

    // epilogue: bias + optional relu, dual-precision store
#pragma unroll
    for (int ni = 0; ni < 2; ++ni) {
        const int gc = gn0 + wn + ni * 16 + l16;
        const float bv = bias ? bias[gc] : 0.f;
#pragma unroll
        for (int mi = 0; mi < 4; ++mi) {
#pragma unroll
            for (int r = 0; r < 8; ++r) {
                const int gr = gm0 + wm + mi * 16 + half * 8 + r;
                float v = acc[mi][ni][r] + bv;
                if (relu) v = fmaxf(v, 0.f);
                if (out16) out16[(size_t)gr * N + gc] = (h16)v;
                if (out32) out32[(size_t)gr * N + gc] = v;
            }
        }
    }
}

// ---------------------------------------------------------------------------
// V transpose: V[b,s,d] (f16, [16384,512]) -> Vt[b,d,s] ([16,512,1024])
// ---------------------------------------------------------------------------
__global__ __launch_bounds__(256) void transpose_v(const h16* __restrict__ V, h16* __restrict__ Vt) {
    __shared__ __align__(16) h16 t[64][72];
    const int b  = blockIdx.z;
    const int s0 = blockIdx.x * 64;
    const int d0 = blockIdx.y * 64;
    const int tid = threadIdx.x;
    const int r = tid >> 2, q = tid & 3;

    const h16* in = V + (size_t)(b * SEQ + s0 + r) * HID + d0 + q * 16;
    *(v8h*)&t[r][q * 16]     = *(const v8h*)in;
    *(v8h*)&t[r][q * 16 + 8] = *(const v8h*)(in + 8);
    __syncthreads();

    v8h o0, o1;
#pragma unroll
    for (int j = 0; j < 8; ++j) { o0[j] = t[q * 16 + j][r]; o1[j] = t[q * 16 + 8 + j][r]; }
    h16* out = Vt + (size_t)(b * HID + d0 + r) * SEQ + s0 + q * 16;
    *(v8h*)out       = o0;
    *(v8h*)(out + 8) = o1;
}

// ---------------------------------------------------------------------------
// Flash attention: 1 wave per (b, head, 16-query tile); 32 keys per iteration.
// Scores: 4x WMMA (2 key-tiles x K=64 in 2 chunks); online softmax via
// half-wave shfl_xor reductions; P bounced through LDS (C-layout -> A-layout);
// PV: 4x WMMA over dh=64.
// ---------------------------------------------------------------------------
#define PTILE 384   // 16 rows * pitch 24 halves
__global__ __launch_bounds__(32) void flash_attn(
    const h16* __restrict__ Q, const h16* __restrict__ Km,
    const h16* __restrict__ Vt, h16* __restrict__ ctx)
{
    __shared__ __align__(16) h16 sp[2 * PTILE];

    const int lane = threadIdx.x & 31;
    const int half = lane >> 4;
    const int l16  = lane & 15;

    const int qt = blockIdx.x & 63;
    const int hh = (blockIdx.x >> 6) & 7;
    const int b  = blockIdx.x >> 9;
    const int m0 = b * SEQ + qt * 16;   // query row in [B*T, HID]
    const int hc = hh * DH;             // head column base

    // Q A-fragments for the two K=32 chunks of dh=64
    v16h qf[2];
#pragma unroll
    for (int c = 0; c < 2; ++c) {
        const h16* p = Q + (size_t)(m0 + l16) * HID + hc + c * 32 + half * 8;
        qf[c] = cat16(*(const v8h*)p, *(const v8h*)(p + 16));
    }

    v8f cacc[4] = {};
    float mr[8], lr[8];
#pragma unroll
    for (int r = 0; r < 8; ++r) { mr[r] = -3.0e38f; lr[r] = 0.f; }

    for (int s0 = 0; s0 < SEQ; s0 += 32) {
        v8f sc0 = {}, sc1 = {};
#pragma unroll
        for (int c = 0; c < 2; ++c) {
            const h16* kb = Km + (size_t)(b * SEQ + s0 + l16) * HID + hc + c * 32 + half * 8;
            v16h kf0 = cat16(*(const v8h*)kb, *(const v8h*)(kb + 16));
            const h16* kb1 = kb + (size_t)16 * HID;
            v16h kf1 = cat16(*(const v8h*)kb1, *(const v8h*)(kb1 + 16));
            sc0 = wmma_f16(qf[c], kf0, sc0);
            sc1 = wmma_f16(qf[c], kf1, sc1);
        }

        // online softmax; each (r, half) pair is one query row
#pragma unroll
        for (int r = 0; r < 8; ++r) {
            float a0 = sc0[r] * 0.125f;   // 1/sqrt(64)
            float a1 = sc1[r] * 0.125f;
            float mx = fmaxf(a0, a1);
#pragma unroll
            for (int msk = 8; msk >= 1; msk >>= 1) mx = fmaxf(mx, __shfl_xor(mx, msk, 32));
            const float mn = fmaxf(mr[r], mx);
            const float sca = __expf(mr[r] - mn);
            mr[r] = mn;
            const float p0 = __expf(a0 - mn);
            const float p1 = __expf(a1 - mn);
            float rs = p0 + p1;
#pragma unroll
            for (int msk = 8; msk >= 1; msk >>= 1) rs += __shfl_xor(rs, msk, 32);
            lr[r] = lr[r] * sca + rs;
#pragma unroll
            for (int nt = 0; nt < 4; ++nt) cacc[nt][r] *= sca;
            const int row = half * 8 + r;
            sp[row * 24 + l16]         = (h16)p0;
            sp[PTILE + row * 24 + l16] = (h16)p1;
        }
        __syncthreads();

        // P as A-matrix (16 queries x 32 keys)
        const h16* pp = sp + l16 * 24 + half * 8;
        v16h pf = cat16(*(const v8h*)pp, *(const v8h*)(pp + PTILE));

        // ctx += P @ V  (V B-fragments contiguous from Vt)
#pragma unroll
        for (int nt = 0; nt < 4; ++nt) {
            const h16* vb = Vt + (size_t)(b * HID + hc + nt * 16 + l16) * SEQ + s0 + half * 8;
            v16h vf = cat16(*(const v8h*)vb, *(const v8h*)(vb + 16));
            cacc[nt] = wmma_f16(pf, vf, cacc[nt]);
        }
        __syncthreads();
    }

    // normalize + store (heads interleaved: D index = h*64 + d)
#pragma unroll
    for (int nt = 0; nt < 4; ++nt) {
        const int gc = hc + nt * 16 + l16;
#pragma unroll
        for (int r = 0; r < 8; ++r) {
            const int gr = m0 + half * 8 + r;
            ctx[(size_t)gr * HID + gc] = (h16)(cacc[nt][r] / lr[r]);
        }
    }
}

// ---------------------------------------------------------------------------
// Fused residual add + LayerNorm (per row of 512), dual-precision output
// ---------------------------------------------------------------------------
__global__ __launch_bounds__(256) void ln_residual(
    const float* __restrict__ res, const float* __restrict__ y,
    const float* __restrict__ g, const float* __restrict__ bta,
    float* __restrict__ o32, h16* __restrict__ o16)
{
    __shared__ float red[256];
    const int row = blockIdx.x, tid = threadIdx.x;
    const size_t base = (size_t)row * HID;

    const float v0 = res[base + tid]       + y[base + tid];
    const float v1 = res[base + 256 + tid] + y[base + 256 + tid];

    red[tid] = v0 + v1;
    __syncthreads();
    for (int s = 128; s > 0; s >>= 1) { if (tid < s) red[tid] += red[tid + s]; __syncthreads(); }
    const float mu = red[0] * (1.f / HID);
    __syncthreads();
    red[tid] = v0 * v0 + v1 * v1;
    __syncthreads();
    for (int s = 128; s > 0; s >>= 1) { if (tid < s) red[tid] += red[tid + s]; __syncthreads(); }
    const float var = red[0] * (1.f / HID) - mu * mu;
    const float inv = rsqrtf(var + 1e-5f);

    const float o0 = (v0 - mu) * inv * g[tid]       + bta[tid];
    const float o1 = (v1 - mu) * inv * g[256 + tid] + bta[256 + tid];
    o32[base + tid] = o0;
    o32[base + 256 + tid] = o1;
    if (o16) { o16[base + tid] = (h16)o0; o16[base + 256 + tid] = (h16)o1; }
}

// ---------------------------------------------------------------------------
// Norm-weighted softmax pooling over the sequence: one block per batch
// ---------------------------------------------------------------------------
__global__ __launch_bounds__(256) void pool_kernel(const float* __restrict__ x, float* __restrict__ pooled) {
    __shared__ float wn[SEQ];
    __shared__ float red[256];
    const int b = blockIdx.x, tid = threadIdx.x;

    for (int t = tid; t < SEQ; t += 256) {
        const float* p = x + (size_t)(b * SEQ + t) * HID;
        float s = 0.f;
        for (int d = 0; d < HID; d += 4) {
            float4 v = *(const float4*)(p + d);
            s += v.x * v.x + v.y * v.y + v.z * v.z + v.w * v.w;
        }
        wn[t] = sqrtf(s);
    }
    __syncthreads();

    float m = -3.0e38f;
    for (int t = tid; t < SEQ; t += 256) m = fmaxf(m, wn[t]);
    red[tid] = m; __syncthreads();
    for (int s = 128; s > 0; s >>= 1) { if (tid < s) red[tid] = fmaxf(red[tid], red[tid + s]); __syncthreads(); }
    const float mx = red[0];
    __syncthreads();

    float s = 0.f;
    for (int t = tid; t < SEQ; t += 256) { float e = __expf(wn[t] - mx); wn[t] = e; s += e; }
    red[tid] = s; __syncthreads();
    for (int st = 128; st > 0; st >>= 1) { if (tid < st) red[tid] += red[tid + st]; __syncthreads(); }
    const float inv = 1.f / red[0];
    __syncthreads();

#pragma unroll
    for (int cc = 0; cc < 2; ++cc) {
        const int c = tid + cc * 256;
        float acc = 0.f;
        for (int t = 0; t < SEQ; ++t) acc += wn[t] * x[(size_t)(b * SEQ + t) * HID + c];
        pooled[(size_t)b * HID + c] = acc * inv;
    }
}

// ---------------------------------------------------------------------------
// Classification head: pooled[16,512] -> relu fc1 (256) -> fc2 (2)
// ---------------------------------------------------------------------------
__global__ __launch_bounds__(256) void head_kernel(
    const float* __restrict__ pooled,
    const float* __restrict__ w1, const float* __restrict__ b1,
    const float* __restrict__ w2, const float* __restrict__ b2,
    float* __restrict__ out)
{
    __shared__ float h1[256];
    const int b = blockIdx.x, j = threadIdx.x;
    float a = b1[j];
    const float* p = pooled + (size_t)b * HID;
    const float* w = w1 + (size_t)j * HID;
    for (int k = 0; k < HID; ++k) a += p[k] * w[k];
    h1[j] = fmaxf(a, 0.f);
    __syncthreads();
    if (j < 2) {
        float o = b2[j];
        const float* w2r = w2 + (size_t)j * 256;
        for (int k = 0; k < 256; ++k) o += h1[k] * w2r[k];
        out[(size_t)b * 2 + j] = o;
    }
}

// ---------------------------------------------------------------------------
// Workspace layout (bytes), with reuse: peak ~275 MB
// ---------------------------------------------------------------------------
static constexpr size_t S_A16  = (size_t)ROWS * HID * 2;    // 16 MB
static constexpr size_t S_A32  = (size_t)ROWS * HID * 4;    // 32 MB
static constexpr size_t S_W16  = (size_t)HID * HID * 2;
static constexpr size_t S_PF16 = (size_t)PF * HID * 2;

static constexpr size_t O_TRG16 = 0;
static constexpr size_t O_SRC16 = O_TRG16 + S_A16;
static constexpr size_t O_WQ16  = O_SRC16 + S_A16;
static constexpr size_t O_WK16  = O_WQ16 + S_W16;
static constexpr size_t O_WV16  = O_WK16 + S_W16;
static constexpr size_t O_WO16  = O_WV16 + S_W16;
static constexpr size_t O_PF1   = O_WO16 + S_W16;
static constexpr size_t O_PF2   = O_PF1 + S_PF16;
static constexpr size_t O_Q16   = O_PF2 + S_PF16;
static constexpr size_t O_K16   = O_Q16 + S_A16;
static constexpr size_t O_V16   = O_K16 + S_A16;
static constexpr size_t O_VT16  = O_V16 + S_A16;
static constexpr size_t O_CTX16 = O_VT16 + S_A16;
static constexpr size_t O_SA32  = O_CTX16 + S_A16;
static constexpr size_t O_X32   = O_SA32 + S_A32;
static constexpr size_t O_X16   = O_X32 + S_A32;
static constexpr size_t O_H16   = O_X16 + S_A16;            // 64 MB
static constexpr size_t O_FF32  = O_SA32;                   // reuse (sa dead)
static constexpr size_t O_X2_32 = O_Q16;                    // reuse (Q/K dead)
static constexpr size_t O_POOL  = O_V16;                    // reuse (V dead)

extern "C" void kernel_launch(void* const* d_in, const int* in_sizes, int n_in,
                              void* d_out, int out_size, void* d_ws, size_t ws_size,
                              hipStream_t stream)
{
    (void)in_sizes; (void)n_in; (void)out_size; (void)ws_size;

    const float* trg   = (const float*)d_in[0];
    const float* src   = (const float*)d_in[1];
    const float* ln_g  = (const float*)d_in[2];
    const float* ln_b  = (const float*)d_in[3];
    const float* wq    = (const float*)d_in[4];
    const float* bq    = (const float*)d_in[5];
    const float* wk    = (const float*)d_in[6];
    const float* bk    = (const float*)d_in[7];
    const float* wv    = (const float*)d_in[8];
    const float* bv    = (const float*)d_in[9];
    const float* wo    = (const float*)d_in[10];
    const float* bo    = (const float*)d_in[11];
    const float* pf1_w = (const float*)d_in[12];
    const float* pf1_b = (const float*)d_in[13];
    const float* pf2_w = (const float*)d_in[14];
    const float* pf2_b = (const float*)d_in[15];
    const float* fc1_w = (const float*)d_in[16];
    const float* fc1_b = (const float*)d_in[17];
    const float* fc2_w = (const float*)d_in[18];
    const float* fc2_b = (const float*)d_in[19];

    char* ws = (char*)d_ws;
    h16*   trg16 = (h16*)(ws + O_TRG16);
    h16*   src16 = (h16*)(ws + O_SRC16);
    h16*   wq16  = (h16*)(ws + O_WQ16);
    h16*   wk16  = (h16*)(ws + O_WK16);
    h16*   wv16  = (h16*)(ws + O_WV16);
    h16*   wo16  = (h16*)(ws + O_WO16);
    h16*   pf116 = (h16*)(ws + O_PF1);
    h16*   pf216 = (h16*)(ws + O_PF2);
    h16*   Q16   = (h16*)(ws + O_Q16);
    h16*   K16   = (h16*)(ws + O_K16);
    h16*   V16   = (h16*)(ws + O_V16);
    h16*   Vt16  = (h16*)(ws + O_VT16);
    h16*   ctx16 = (h16*)(ws + O_CTX16);
    float* sa32  = (float*)(ws + O_SA32);
    float* x32   = (float*)(ws + O_X32);
    h16*   x16   = (h16*)(ws + O_X16);
    h16*   h16b  = (h16*)(ws + O_H16);
    float* ff32  = (float*)(ws + O_FF32);
    float* x232  = (float*)(ws + O_X2_32);
    float* pool  = (float*)(ws + O_POOL);
    float* out   = (float*)d_out;

    // 1) convert activations + weights to f16
    cvt_f32_f16<<<2048, 256, 0, stream>>>(trg,   trg16, (size_t)ROWS * HID);
    cvt_f32_f16<<<2048, 256, 0, stream>>>(src,   src16, (size_t)ROWS * HID);
    cvt_f32_f16<<<512,  256, 0, stream>>>(wq,    wq16,  (size_t)HID * HID);
    cvt_f32_f16<<<512,  256, 0, stream>>>(wk,    wk16,  (size_t)HID * HID);
    cvt_f32_f16<<<512,  256, 0, stream>>>(wv,    wv16,  (size_t)HID * HID);
    cvt_f32_f16<<<512,  256, 0, stream>>>(wo,    wo16,  (size_t)HID * HID);
    cvt_f32_f16<<<1024, 256, 0, stream>>>(pf1_w, pf116, (size_t)PF * HID);
    cvt_f32_f16<<<1024, 256, 0, stream>>>(pf2_w, pf216, (size_t)HID * PF);

    const dim3 g512(HID / GBM, ROWS / GBM);   // (4, 128)
    const dim3 gPF(PF / GBM, ROWS / GBM);     // (16, 128)

    // 2) Q/K/V projections (f16 out)
    wmma_gemm<<<g512, 256, 0, stream>>>(trg16, wq16, bq, Q16, nullptr, ROWS, HID, HID, 0);
    wmma_gemm<<<g512, 256, 0, stream>>>(src16, wk16, bk, K16, nullptr, ROWS, HID, HID, 0);
    wmma_gemm<<<g512, 256, 0, stream>>>(src16, wv16, bv, V16, nullptr, ROWS, HID, HID, 0);

    // 3) V transpose for contiguous B-fragments in attention
    transpose_v<<<dim3(SEQ / 64, HID / 64, NB), 256, 0, stream>>>(V16, Vt16);

    // 4) flash attention: B*H*(T/16) = 8192 single-wave blocks
    flash_attn<<<NB * HEADS * (SEQ / 16), 32, 0, stream>>>(Q16, K16, Vt16, ctx16);

    // 5) output projection (f32 out) + LN1 (residual = trg)
    wmma_gemm<<<g512, 256, 0, stream>>>(ctx16, wo16, bo, nullptr, sa32, ROWS, HID, HID, 0);
    ln_residual<<<ROWS, 256, 0, stream>>>(trg, sa32, ln_g, ln_b, x32, x16);

    // 6) FFN: relu(x @ pf1^T) -> f16 ; @ pf2^T -> f32 ; LN2
    wmma_gemm<<<gPF, 256, 0, stream>>>(x16, pf116, pf1_b, h16b, nullptr, ROWS, PF, HID, 1);
    wmma_gemm<<<g512, 256, 0, stream>>>(h16b, pf216, pf2_b, nullptr, ff32, ROWS, HID, PF, 0);
    ln_residual<<<ROWS, 256, 0, stream>>>(x32, ff32, ln_g, ln_b, x232, nullptr);

    // 7) norm-weighted pooling + classification head
    pool_kernel<<<NB, 256, 0, stream>>>(x232, pool);
    head_kernel<<<NB, 256, 0, stream>>>(pool, fc1_w, fc1_b, fc2_w, fc2_b, out);
}
